// SetAbstraction_42812234007147
// MI455X (gfx1250) — compile-verified, matrix operand order checked by reading, not run on hardware
//
#include <hip/hip_runtime.h>
#include <hip/hip_bf16.h>

// ---------------------------------------------------------------------------
// PointNet++ SetAbstraction for MI455X (gfx1250, wave32, WMMA)
//   B=8, N=8192, NPOINT=2048, NSAMPLE=32, C_in=64, mlp=[64,64,128]
// Pipeline:
//   K1 fps_kernel    : 1 block/batch, serial FPS, writes new_xyz (ws + out)
//   K2 ballq_kernel  : 1 wave/centroid, ball query + gather -> f16 feat rows
//                      [B*S*32, 96] (67 valid channels, zero padded)
//   K3 mlp_kernel    : 1 wave/centroid, 3-layer MLP via v_wmma_f32_16x16x32_f16
//                      + BN(folded) + ReLU + max over nsample, writes new_points
// ---------------------------------------------------------------------------

typedef __attribute__((ext_vector_type(16))) _Float16 v16h;
typedef __attribute__((ext_vector_type(8)))  _Float16 v8h;
typedef __attribute__((ext_vector_type(8)))  float    v8f;

#define BB      8
#define NN      8192
#define SS      2048
#define NSAMP   32
#define KPAD    96     // 67 channels padded to 3 k-tiles of 32
#define RAD2    0.81f  // 0.9^2

// -------------------------------- K1: FPS ---------------------------------

__global__ __launch_bounds__(1024) void fps_kernel(const float* __restrict__ xyz,
                                                   float* __restrict__ newxyz_ws,
                                                   float* __restrict__ out_xyz) {
  const int b   = blockIdx.x;
  const int tid = threadIdx.x;
  const float* xb = xyz + (size_t)b * 3 * NN;

  // 8 points per thread, coalesced: n = j*1024 + tid
  float px[8], py[8], pz[8], dist[8];
#pragma unroll
  for (int j = 0; j < 8; ++j) {
    int n = j * 1024 + tid;
    px[j] = xb[n];
    py[j] = xb[NN + n];
    pz[j] = xb[2 * NN + n];
    dist[j] = 1e10f;
  }

  __shared__ float s_c[3];
  __shared__ float s_val[32];
  __shared__ int   s_idx[32];
  __shared__ int   s_far;

  const int lane = tid & 31;
  const int wid  = tid >> 5;

  int farthest = 0;
  for (int it = 0; it < SS; ++it) {
    // owning thread republishes centroid coords (L2-resident reload)
    if (tid == (farthest & 1023)) {
      s_c[0] = xb[farthest];
      s_c[1] = xb[NN + farthest];
      s_c[2] = xb[2 * NN + farthest];
    }
    __syncthreads();
    const float cx = s_c[0], cy = s_c[1], cz = s_c[2];

    if (tid == 0) {
      newxyz_ws[((size_t)b * SS + it) * 3 + 0] = cx;
      newxyz_ws[((size_t)b * SS + it) * 3 + 1] = cy;
      newxyz_ws[((size_t)b * SS + it) * 3 + 2] = cz;
      out_xyz[(size_t)b * 3 * SS + 0 * SS + it] = cx;
      out_xyz[(size_t)b * 3 * SS + 1 * SS + it] = cy;
      out_xyz[(size_t)b * 3 * SS + 2 * SS + it] = cz;
    }

    // update running min distance, track local argmax
    float best = -1.0f;
    int   bi   = 0x7fffffff;
#pragma unroll
    for (int j = 0; j < 8; ++j) {
      float dx = px[j] - cx, dy = py[j] - cy, dz = pz[j] - cz;
      float d = dx * dx + dy * dy + dz * dz;
      dist[j] = fminf(dist[j], d);
      int n = j * 1024 + tid;
      if (dist[j] > best || (dist[j] == best && n < bi)) { best = dist[j]; bi = n; }
    }
    // wave32 argmax reduction (first-max tie break)
#pragma unroll
    for (int off = 16; off > 0; off >>= 1) {
      float ov = __shfl_xor(best, off, 32);
      int   oi = __shfl_xor(bi, off, 32);
      if (ov > best || (ov == best && oi < bi)) { best = ov; bi = oi; }
    }
    if (lane == 0) { s_val[wid] = best; s_idx[wid] = bi; }
    __syncthreads();
    if (wid == 0) {
      float v = s_val[lane];
      int   i2 = s_idx[lane];
#pragma unroll
      for (int off = 16; off > 0; off >>= 1) {
        float ov = __shfl_xor(v, off, 32);
        int   oi = __shfl_xor(i2, off, 32);
        if (ov > v || (ov == v && oi < i2)) { v = ov; i2 = oi; }
      }
      if (lane == 0) s_far = i2;
    }
    __syncthreads();
    farthest = s_far;
  }
}

// -------------------------- K2: ball query + gather ------------------------

__global__ __launch_bounds__(256) void ballq_kernel(const float* __restrict__ xyz,
                                                    const float* __restrict__ points,
                                                    const float* __restrict__ newxyz_ws,
                                                    _Float16* __restrict__ feat) {
  const int wid  = threadIdx.x >> 5;
  const int lane = threadIdx.x & 31;
  const int bid  = blockIdx.x;           // B * S/8 blocks
  const int b    = bid / (SS / 8);
  const int s    = (bid % (SS / 8)) * 8 + wid;

  __shared__ int s_gidx[8][NSAMP];

  const float cx = newxyz_ws[((size_t)b * SS + s) * 3 + 0];
  const float cy = newxyz_ws[((size_t)b * SS + s) * 3 + 1];
  const float cz = newxyz_ws[((size_t)b * SS + s) * 3 + 2];

  const float* xb = xyz + (size_t)b * 3 * NN;

  int cnt = 0;  // uniform across the wave (ballot-derived)
  for (int base = 0; base < NN; base += 32) {
    int n = base + lane;
    float dx = xb[n] - cx;
    float dy = xb[NN + n] - cy;
    float dz = xb[2 * NN + n] - cz;
    float d2 = dx * dx + dy * dy + dz * dz;
    bool inball = (d2 <= RAD2);
    unsigned mask = (unsigned)__ballot(inball);
    int pos = cnt + __popc(mask & ((1u << lane) - 1u));
    if (inball && pos < NSAMP) s_gidx[wid][pos] = n;
    cnt += __popc(mask);
    if (cnt >= NSAMP) break;
  }
  if (cnt > NSAMP) cnt = NSAMP;

  int first = (cnt > 0) ? s_gidx[wid][0] : 0;
  int myidx = (lane < cnt) ? s_gidx[wid][lane] : first;

  // gather & emit one f16 feature row per sample (lane = sample)
  const size_t row = ((size_t)(b * SS + s)) * NSAMP + lane;
  _Float16* fr = feat + row * KPAD;
  fr[0] = (_Float16)(xb[myidx] - cx);
  fr[1] = (_Float16)(xb[NN + myidx] - cy);
  fr[2] = (_Float16)(xb[2 * NN + myidx] - cz);
  const float* pb = points + (size_t)b * 64 * NN;
#pragma unroll 8
  for (int c = 0; c < 64; ++c) fr[3 + c] = (_Float16)pb[(size_t)c * NN + myidx];
#pragma unroll
  for (int c = 67; c < KPAD; ++c) fr[c] = (_Float16)0.0f;
}

// ------------------------------ K3: WMMA MLP -------------------------------

// Build an A/B-style v16h fragment: halves[0..7] at p, halves[8..15] at p+16
__device__ __forceinline__ v16h load_frag(const _Float16* p) {
  const v8h lo = *(const v8h*)p;
  const v8h hi = *(const v8h*)(p + 16);
  v16h r;
#pragma unroll
  for (int i = 0; i < 8; ++i) { r[i] = lo[i]; r[8 + i] = hi[i]; }
  return r;
}

__global__ __launch_bounds__(256) void mlp_kernel(const _Float16* __restrict__ feat,
                                                  const float* __restrict__ w1, const float* __restrict__ b1,
                                                  const float* __restrict__ g1, const float* __restrict__ be1,
                                                  const float* __restrict__ w2, const float* __restrict__ b2,
                                                  const float* __restrict__ g2, const float* __restrict__ be2,
                                                  const float* __restrict__ w3, const float* __restrict__ b3,
                                                  const float* __restrict__ g3, const float* __restrict__ be3,
                                                  float* __restrict__ out_pts) {
  __shared__ __attribute__((aligned(16))) _Float16 s_w1[64][KPAD]; // 12KB (k padded)
  __shared__ __attribute__((aligned(16))) _Float16 s_w2[64][64];   // 8KB
  __shared__ __attribute__((aligned(16))) _Float16 s_w3[128][64];  // 16KB
  __shared__ __attribute__((aligned(16))) _Float16 s_h[8][16][64]; // 16KB (per-wave act tile)
  __shared__ float s_a1[64], s_b1[64], s_a2[64], s_b2[64], s_a3[128], s_b3[128];

  const int tid = threadIdx.x;
  // stage weights (f32 -> f16) + folded BN params
  for (int i = tid; i < 64 * KPAD; i += 256) {
    int o = i / KPAD, c = i % KPAD;
    s_w1[o][c] = (c < 67) ? (_Float16)w1[o * 67 + c] : (_Float16)0.0f;
  }
  for (int i = tid; i < 64 * 64; i += 256)  s_w2[i / 64][i % 64] = (_Float16)w2[i];
  for (int i = tid; i < 128 * 64; i += 256) s_w3[i / 64][i % 64] = (_Float16)w3[i];
  const float inv = rsqrtf(1.0f + 1e-5f);
  for (int i = tid; i < 64; i += 256) {
    float a1 = g1[i] * inv; s_a1[i] = a1; s_b1[i] = b1[i] * a1 + be1[i];
    float a2 = g2[i] * inv; s_a2[i] = a2; s_b2[i] = b2[i] * a2 + be2[i];
  }
  for (int i = tid; i < 128; i += 256) {
    float a3 = g3[i] * inv; s_a3[i] = a3; s_b3[i] = b3[i] * a3 + be3[i];
  }
  __syncthreads();

  const int wid  = tid >> 5;
  const int lane = tid & 31;
  const int b    = blockIdx.x / (SS / 8);
  const int s    = (blockIdx.x % (SS / 8)) * 8 + wid;
  const size_t rowbase = ((size_t)(b * SS + s)) * NSAMP;

  const int half16 = lane >> 4;        // 0: lanes 0-15, 1: lanes 16-31
  const int khalf  = half16 * 8;       // K sub-offset within a 32-wide k-tile
  const int mlane  = lane & 15;        // row/col within tile
  const int rbase  = half16 * 8;       // C/D row offset for this half-wave

  float maxv[8];
#pragma unroll
  for (int nt = 0; nt < 8; ++nt) maxv[nt] = -1e30f;

  for (int strip = 0; strip < 2; ++strip) {
    const _Float16* arow = feat + (rowbase + strip * 16 + mlane) * KPAD;
    __builtin_prefetch(arow, 0, 3);                    // global_prefetch_b8
    if (strip == 0) __builtin_prefetch(arow + 16 * KPAD, 0, 3);

    // ---- layer 1: 96(67) -> 64 ----
#pragma unroll
    for (int nt = 0; nt < 4; ++nt) {
      v8f acc = {};
      const _Float16* brow = &s_w1[nt * 16 + mlane][0];
#pragma unroll
      for (int kt = 0; kt < 3; ++kt) {
        v16h A = load_frag(arow + kt * 32 + khalf);
        v16h Bf = load_frag(brow + kt * 32 + khalf);
        acc = __builtin_amdgcn_wmma_f32_16x16x32_f16(false, A, false, Bf,
                                                     (short)0, acc, false, false);
      }
      const int ch = nt * 16 + mlane;
      const float al = s_a1[ch], bt = s_b1[ch];
#pragma unroll
      for (int r = 0; r < 8; ++r) {
        float v = fmaxf(acc[r] * al + bt, 0.0f);
        s_h[wid][rbase + r][ch] = (_Float16)v;
      }
    }

    // ---- layer 2: 64 -> 64 (A-frags preloaded so s_h can be reused) ----
    {
      const _Float16* hrow = &s_h[wid][mlane][0];
      v16h A0 = load_frag(hrow + khalf);
      v16h A1 = load_frag(hrow + 32 + khalf);
#pragma unroll
      for (int nt = 0; nt < 4; ++nt) {
        v8f acc = {};
        const _Float16* brow = &s_w2[nt * 16 + mlane][0];
        v16h B0 = load_frag(brow + khalf);
        v16h B1 = load_frag(brow + 32 + khalf);
        acc = __builtin_amdgcn_wmma_f32_16x16x32_f16(false, A0, false, B0,
                                                     (short)0, acc, false, false);
        acc = __builtin_amdgcn_wmma_f32_16x16x32_f16(false, A1, false, B1,
                                                     (short)0, acc, false, false);
        const int ch = nt * 16 + mlane;
        const float al = s_a2[ch], bt = s_b2[ch];
#pragma unroll
        for (int r = 0; r < 8; ++r) {
          float v = fmaxf(acc[r] * al + bt, 0.0f);
          s_h[wid][rbase + r][ch] = (_Float16)v;
        }
      }
    }

    // ---- layer 3: 64 -> 128, fused ReLU + max over rows ----
    {
      const _Float16* hrow = &s_h[wid][mlane][0];
      v16h A0 = load_frag(hrow + khalf);
      v16h A1 = load_frag(hrow + 32 + khalf);
#pragma unroll
      for (int nt = 0; nt < 8; ++nt) {
        v8f acc = {};
        const _Float16* brow = &s_w3[nt * 16 + mlane][0];
        v16h B0 = load_frag(brow + khalf);
        v16h B1 = load_frag(brow + 32 + khalf);
        acc = __builtin_amdgcn_wmma_f32_16x16x32_f16(false, A0, false, B0,
                                                     (short)0, acc, false, false);
        acc = __builtin_amdgcn_wmma_f32_16x16x32_f16(false, A1, false, B1,
                                                     (short)0, acc, false, false);
        const int ch = nt * 16 + mlane;
        const float al = s_a3[ch], bt = s_b3[ch];
        float m = -1e30f;
#pragma unroll
        for (int r = 0; r < 8; ++r) {
          float v = fmaxf(acc[r] * al + bt, 0.0f);
          m = fmaxf(m, v);
        }
        m = fmaxf(m, __shfl_xor(m, 16, 32));   // merge rows 0-7 / 8-15
        maxv[nt] = fmaxf(maxv[nt], m);
      }
    }
  }

  // lanes 0..15 hold per-channel maxima (channel = nt*16 + lane)
  if (lane < 16) {
#pragma unroll
    for (int nt = 0; nt < 8; ++nt) {
      int ch = nt * 16 + lane;
      out_pts[(size_t)b * 128 * SS + (size_t)ch * SS + s] = maxv[nt];
    }
  }
}

// ------------------------------- launcher ----------------------------------

extern "C" void kernel_launch(void* const* d_in, const int* in_sizes, int n_in,
                              void* d_out, int out_size, void* d_ws, size_t ws_size,
                              hipStream_t stream) {
  (void)in_sizes; (void)n_in; (void)out_size; (void)ws_size;

  const float* xyz    = (const float*)d_in[0];
  const float* points = (const float*)d_in[1];
  const float* w1 = (const float*)d_in[2];
  const float* b1 = (const float*)d_in[3];
  const float* g1 = (const float*)d_in[4];
  const float* be1 = (const float*)d_in[5];
  const float* w2 = (const float*)d_in[6];
  const float* b2 = (const float*)d_in[7];
  const float* g2 = (const float*)d_in[8];
  const float* be2 = (const float*)d_in[9];
  const float* w3 = (const float*)d_in[10];
  const float* b3 = (const float*)d_in[11];
  const float* g3 = (const float*)d_in[12];
  const float* be3 = (const float*)d_in[13];

  float* out = (float*)d_out;
  float* out_xyz = out;                              // [B,3,S]
  float* out_pts = out + (size_t)BB * 3 * SS;        // [B,128,S]

  float* ws_newxyz = (float*)d_ws;                   // [B,S,3] f32
  _Float16* ws_feat = (_Float16*)((char*)d_ws + (size_t)BB * SS * 3 * sizeof(float));

  fps_kernel<<<BB, 1024, 0, stream>>>(xyz, ws_newxyz, out_xyz);
  ballq_kernel<<<BB * (SS / 8), 256, 0, stream>>>(xyz, points, ws_newxyz, ws_feat);
  mlp_kernel<<<BB * (SS / 8), 256, 0, stream>>>(ws_feat,
                                                w1, b1, g1, be1,
                                                w2, b2, g2, be2,
                                                w3, b3, g3, be3,
                                                out_pts);
}
